// GATModel_19155554140458
// MI455X (gfx1250) — matrix operand order repaired and checked by reading, not compiled, not run on hardware
//
#include <hip/hip_runtime.h>
#include <hip/hip_bf16.h>
#include <math.h>

typedef __attribute__((ext_vector_type(16))) _Float16 v16h;
typedef __attribute__((ext_vector_type(8)))  float    v8f;

#define NEG_SLOPE 0.2f

__device__ __forceinline__ void atomic_max_f32(float* addr, float v) {
    // IEEE ordering trick: non-negative floats compare as signed ints,
    // negative floats compare reversed as unsigned ints.
    if (v >= 0.0f)
        atomicMax((int*)addr, __float_as_int(v));
    else
        atomicMin((unsigned int*)addr, __float_as_uint(v));
}

// ---------------------------------------------------------------------------
// Pack B[K,Nc] (f32, row-major) into per-fragment WMMA B layout, f16:
//   Bpk[kg][nt][lane][hv] , kg = k/32, nt = col/16, 16 halves (32B) per lane.
// Matches V_WMMA_F32_16X16X32_F16 B-fragment: col = nt*16 + (lane&15),
//   half hv holds K = kg*32 + 16*(hv>>3) + 8*(lane>>4) + (hv&7).
// ---------------------------------------------------------------------------
__global__ void k_pack_b(const float* __restrict__ B, _Float16* __restrict__ Bpk,
                         int K, int Nc) {
    int t = blockIdx.x * blockDim.x + threadIdx.x;     // one thread per half
    int tilesN = Nc >> 4;
    int total = (K >> 5) * tilesN * 32 * 16;
    if (t >= total) return;
    int hv   = t & 15;
    int lane = (t >> 4) & 31;
    int nt   = (t >> 9) % tilesN;
    int kg   = (t >> 9) / tilesN;
    int grp = lane >> 4, l15 = lane & 15;
    int k   = (kg << 5) + ((hv & 8) << 1) + (grp << 3) + (hv & 7);
    int col = (nt << 4) + l15;
    Bpk[t] = (_Float16)B[(size_t)k * Nc + col];
}

// ---------------------------------------------------------------------------
// C[M,Nc] = A[M,K] * Bpk; A f32 in memory (b64 pair loads + cvt), B prepacked
// f16 fragments (one aligned 32B load each, no conversion, no per-frag wait).
// One wave computes a 16 x (COLS*16) stripe: A fragment converted once per
// K-step, reused by COLS back-to-back WMMAs.
// ---------------------------------------------------------------------------
template <int COLS>
__global__ void k_gemm_wmma(const float* __restrict__ A,
                            const _Float16* __restrict__ Bpk,
                            float* __restrict__ C,
                            int M, int K, int Nc) {
    const int lane = threadIdx.x & 31;
    const int wid  = blockIdx.x * (blockDim.x >> 5) + (threadIdx.x >> 5);
    const int tilesM    = M >> 4;
    const int tilesN    = Nc >> 4;
    const int colGroups = tilesN / COLS;
    if (wid >= tilesM * colGroups) return;         // wave-uniform exit
    const int tileM = (wid / colGroups) << 4;
    const int nt0   = (wid % colGroups) * COLS;

    const int grp = lane >> 4;                     // 0 or 1
    const int l15 = lane & 15;

    v8f acc[COLS];
    #pragma unroll
    for (int c = 0; c < COLS; ++c) acc[c] = (v8f){};

    const float2* arow2 = (const float2*)(A + (size_t)(tileM + l15) * K);
    const v16h*   Bf    = (const v16h*)Bpk;        // 32B fragments, one per lane

    #pragma unroll 2
    for (int k0 = 0; k0 < K; k0 += 32) {
        // A frag (16x32, MxK): VGPR v in 0..3 -> K = k0 + 8*grp + 2v{,+1};
        //                      VGPR v in 4..7 -> +16  (pairs -> b64 loads)
        v16h af;
        #pragma unroll
        for (int v = 0; v < 8; ++v) {
            int k = k0 + ((v & 4) ? 16 : 0) + (grp << 3) + ((v & 3) << 1);
            float2 p = arow2[k >> 1];
            af[2 * v]     = (_Float16)p.x;
            af[2 * v + 1] = (_Float16)p.y;
        }
        const size_t fb = ((size_t)(k0 >> 5) * tilesN + nt0) * 32 + lane;
        #pragma unroll
        for (int c = 0; c < COLS; ++c) {
            v16h bf = Bf[fb + (size_t)c * 32];
            acc[c] = __builtin_amdgcn_wmma_f32_16x16x32_f16(
                /*neg_a=*/false, af, /*neg_b=*/false, bf,
                /*c_mod=*/(short)0, acc[c], /*reuse_a=*/false, /*reuse_b=*/false);
        }
    }
    // C/D layout: VGPR r -> row = tileM + r + 8*grp, col = 16*(nt0+c) + l15
    #pragma unroll
    for (int c = 0; c < COLS; ++c) {
        float* crow = C + (size_t)(tileM + (grp << 3)) * Nc + ((nt0 + c) << 4) + l15;
        #pragma unroll
        for (int r = 0; r < 8; ++r)
            crow[(size_t)r * Nc] = acc[c][r];
    }
}

// a_src[n,h] = sum_c h[n,h,c]*att_src[h,c];  a_dst likewise.
__global__ void k_att_scores(const float* __restrict__ h,
                             const float* __restrict__ att_src,
                             const float* __restrict__ att_dst,
                             float* __restrict__ a_src,
                             float* __restrict__ a_dst,
                             int N, int H, int C) {
    int t = blockIdx.x * blockDim.x + threadIdx.x;
    if (t >= N * H) return;
    int n = t / H, hd = t % H;
    const float* hp = h + (size_t)n * H * C + (size_t)hd * C;
    const float* as = att_src + (size_t)hd * C;
    const float* ad = att_dst + (size_t)hd * C;
    float s = 0.f, d = 0.f;
    for (int c = 0; c < C; ++c) { float v = hp[c]; s += v * as[c]; d += v * ad[c]; }
    a_src[t] = s;
    a_dst[t] = d;
}

__global__ void k_fill_neg_inf(float* __restrict__ p, int n) {
    int t = blockIdx.x * blockDim.x + threadIdx.x;
    if (t < n) p[t] = __int_as_float(0xFF800000);  // -inf
}

// ---- per-edge kernels: one thread per edge, loop over heads ----
__global__ void k_edge_max(const float* __restrict__ a_src,
                           const float* __restrict__ a_dst,
                           const int* __restrict__ src,
                           const int* __restrict__ dst,
                           float* __restrict__ amax, int E, int H) {
    int e = blockIdx.x * blockDim.x + threadIdx.x;
    if (e >= E) return;
    int s = src[e], d = dst[e];
    for (int hd = 0; hd < H; ++hd) {
        float a = a_src[(size_t)s * H + hd] + a_dst[(size_t)d * H + hd];
        a = (a > 0.f) ? a : NEG_SLOPE * a;         // leaky_relu
        atomic_max_f32(&amax[(size_t)d * H + hd], a);
    }
}

__global__ void k_edge_exp(const float* __restrict__ a_src,
                           const float* __restrict__ a_dst,
                           const int* __restrict__ src,
                           const int* __restrict__ dst,
                           const float* __restrict__ amax,
                           float* __restrict__ denom,
                           float* __restrict__ attn, int E, int H) {
    int e = blockIdx.x * blockDim.x + threadIdx.x;
    if (e >= E) return;
    int s = src[e], d = dst[e];
    for (int hd = 0; hd < H; ++hd) {
        float a = a_src[(size_t)s * H + hd] + a_dst[(size_t)d * H + hd];
        a = (a > 0.f) ? a : NEG_SLOPE * a;
        float ex = expf(a - amax[(size_t)d * H + hd]);  // amax finite: d has an edge
        attn[(size_t)e * H + hd] = ex;
        atomicAdd(&denom[(size_t)d * H + hd], ex);
    }
}

__global__ void k_edge_norm(const int* __restrict__ dst,
                            const float* __restrict__ denom,
                            float* __restrict__ attn, int E, int H) {
    int e = blockIdx.x * blockDim.x + threadIdx.x;
    if (e >= E) return;
    int d = dst[e];
    for (int hd = 0; hd < H; ++hd)
        attn[(size_t)e * H + hd] /= (denom[(size_t)d * H + hd] + 1e-16f);
}

// One wave per edge: gather h[src] channels, scale by attn head weight,
// scatter-add into out[dst]. HC = H*C channels, lane-strided (coalesced).
__global__ void k_aggregate(const float* __restrict__ h,
                            const float* __restrict__ attn,
                            const int* __restrict__ src,
                            const int* __restrict__ dst,
                            float* __restrict__ out,
                            int E, int H, int C) {
    int e = blockIdx.x * (blockDim.x >> 5) + (threadIdx.x >> 5);
    if (e >= E) return;                            // wave-uniform exit
    int lane = threadIdx.x & 31;
    int s = src[e], d = dst[e];
    int HC = H * C;
    const float* hs = h + (size_t)s * HC;
    const float* at = attn + (size_t)e * H;
    float* od = out + (size_t)d * HC;
    for (int j = lane; j < HC; j += 32)
        atomicAdd(&od[j], at[j / C] * hs[j]);
}

__global__ void k_elu(float* __restrict__ p, size_t n) {
    size_t t = (size_t)blockIdx.x * blockDim.x + threadIdx.x;
    if (t >= n) return;
    float v = p[t];
    p[t] = (v > 0.f) ? v : (expf(v) - 1.0f);
}

// ---------------------------------------------------------------------------
extern "C" void kernel_launch(void* const* d_in, const int* in_sizes, int n_in,
                              void* d_out, int out_size, void* d_ws, size_t ws_size,
                              hipStream_t stream) {
    const float* x        = (const float*)d_in[0];
    const float* W1       = (const float*)d_in[1];
    const float* att_src1 = (const float*)d_in[2];
    const float* att_dst1 = (const float*)d_in[3];
    const float* W2       = (const float*)d_in[4];
    const float* att_src2 = (const float*)d_in[5];
    const float* att_dst2 = (const float*)d_in[6];
    const int*   eidx     = (const int*)d_in[7];

    const int F_IN = 256, H1 = 8, C1 = 32, HC1 = 256, F_OUT = 64;
    const int N = in_sizes[0] / F_IN;
    const int E = in_sizes[7] / 2;
    const int* srcI = eidx;       // edge_index[0]
    const int* dstI = eidx + E;   // edge_index[1]

    float* ws = (float*)d_ws;
    const size_t N256 = (size_t)N * HC1;
    float* h1    = ws;
    float* hbuf  = h1    + N256;                 // layer-1 aggregate -> ELU (gemm2 input)
    float* ascr1 = hbuf  + N256;
    float* adst1 = ascr1 + (size_t)N * H1;
    float* amax1 = adst1 + (size_t)N * H1;
    float* den1  = amax1 + (size_t)N * H1;
    float* attn1 = den1  + (size_t)N * H1;
    float* h2    = attn1 + (size_t)E * H1;
    float* ascr2 = h2    + (size_t)N * F_OUT;
    float* adst2 = ascr2 + (size_t)N;
    float* amax2 = adst2 + (size_t)N;
    float* den2  = amax2 + (size_t)N;
    float* attn2 = den2  + (size_t)N;
    // packed f16 weights (32B-aligned: all previous region sizes are 32B multiples)
    _Float16* bpk1 = (_Float16*)(attn2 + (size_t)E);          // 256*256 halves
    _Float16* bpk2 = bpk1 + (size_t)F_IN * HC1;               // 256*64  halves

    // zero accumulators (graph-capture-safe memsets on stream)
    hipMemsetAsync(hbuf, 0, N256 * sizeof(float), stream);
    hipMemsetAsync(den1, 0, (size_t)N * H1 * sizeof(float), stream);
    hipMemsetAsync(den2, 0, (size_t)N * sizeof(float), stream);
    hipMemsetAsync(d_out, 0, (size_t)N * F_OUT * sizeof(float), stream);

    // pack both weight matrices into WMMA B-fragment layout (f16)
    {
        int tot1 = (F_IN / 32) * (HC1 / 16) * 512;
        int tot2 = (HC1 / 32) * (F_OUT / 16) * 512;
        k_pack_b<<<(tot1 + 255) / 256, 256, 0, stream>>>(W1, bpk1, F_IN, HC1);
        k_pack_b<<<(tot2 + 255) / 256, 256, 0, stream>>>(W2, bpk2, HC1, F_OUT);
    }

    // ---------------- Layer 1 ----------------
    {
        // 16x128 stripe per wave: tilesN=16 -> 2 column groups of 8
        int waves = (N / 16) * ((HC1 / 16) / 8);
        k_gemm_wmma<8><<<(waves + 3) / 4, 128, 0, stream>>>(x, bpk1, h1, N, F_IN, HC1);
    }
    k_att_scores<<<(N * H1 + 255) / 256, 256, 0, stream>>>(h1, att_src1, att_dst1,
                                                           ascr1, adst1, N, H1, C1);
    k_fill_neg_inf<<<(N * H1 + 255) / 256, 256, 0, stream>>>(amax1, N * H1);
    k_edge_max <<<(E + 255) / 256, 256, 0, stream>>>(ascr1, adst1, srcI, dstI, amax1, E, H1);
    k_edge_exp <<<(E + 255) / 256, 256, 0, stream>>>(ascr1, adst1, srcI, dstI, amax1,
                                                     den1, attn1, E, H1);
    k_edge_norm<<<(E + 255) / 256, 256, 0, stream>>>(dstI, den1, attn1, E, H1);
    k_aggregate<<<(E + 7) / 8, 256, 0, stream>>>(h1, attn1, srcI, dstI, hbuf, E, H1, C1);
    k_elu<<<(unsigned)((N256 + 255) / 256), 256, 0, stream>>>(hbuf, N256);

    // ---------------- Layer 2 ----------------
    {
        // 16x64 stripe per wave: tilesN=4 -> 1 column group of 4
        int waves = (N / 16) * ((F_OUT / 16) / 4);
        k_gemm_wmma<4><<<(waves + 3) / 4, 128, 0, stream>>>(hbuf, bpk2, h2, N, HC1, F_OUT);
    }
    k_att_scores<<<(N + 255) / 256, 256, 0, stream>>>(h2, att_src2, att_dst2,
                                                      ascr2, adst2, N, 1, F_OUT);
    k_fill_neg_inf<<<(N + 255) / 256, 256, 0, stream>>>(amax2, N);
    k_edge_max <<<(E + 255) / 256, 256, 0, stream>>>(ascr2, adst2, srcI, dstI, amax2, E, 1);
    k_edge_exp <<<(E + 255) / 256, 256, 0, stream>>>(ascr2, adst2, srcI, dstI, amax2,
                                                     den2, attn2, E, 1);
    k_edge_norm<<<(E + 255) / 256, 256, 0, stream>>>(dstI, den2, attn2, E, 1);
    // H=1 => out.mean(axis=1) == the single head; aggregate straight into d_out.
    k_aggregate<<<(E + 7) / 8, 256, 0, stream>>>(h2, attn2, srcI, dstI,
                                                 (float*)d_out, E, 1, F_OUT);
}
